// CodeBook_35545149342217
// MI455X (gfx1250) — matrix operand order repaired and checked by reading, not compiled, not run on hardware
//
#include <hip/hip_runtime.h>
#include <hip/hip_bf16.h>

typedef __attribute__((ext_vector_type(16))) __bf16 v16bf;
typedef __attribute__((ext_vector_type(8)))  float  v8f;

#define D_DIM 256
#define K_CB  1024
#define N_ROWS 65536
#define LDS_B_STRIDE 264   // 256 + 8 bf16 pad -> conflict-free b128 DS reads

typedef __attribute__((address_space(3))) void lds_void;

union AFrag { v16bf v; __bf16 e[16]; float4 f4[2]; };
union BFrag { v16bf v; float4 f4[2]; };

// ---------------------------------------------------------------------------
// Kernel 1: transpose embedding [D][K] -> [K][D], split bf16 hi/lo, 0.5*||e||^2
// ---------------------------------------------------------------------------
__global__ void vq_prep_kernel(const float* __restrict__ E,
                               __bf16* __restrict__ eThi,
                               __bf16* __restrict__ eTlo,
                               float*  __restrict__ eTf,
                               float*  __restrict__ cn) {
    int c = blockIdx.x * blockDim.x + threadIdx.x;   // 0..1023, coalesced on c
    if (c >= K_CB) return;
    float s = 0.0f;
    #pragma unroll 4
    for (int d = 0; d < D_DIM; ++d) {
        float x = E[(size_t)d * K_CB + c];
        __bf16 h = (__bf16)x;
        float hf = (float)h;
        __bf16 l = (__bf16)(x - hf);
        eThi[(size_t)c * D_DIM + d] = h;
        eTlo[(size_t)c * D_DIM + d] = l;
        eTf [(size_t)c * D_DIM + d] = x;
        s += x * x;                                   // fp32, fixed order
    }
    cn[c] = 0.5f * s;
}

// ---------------------------------------------------------------------------
// Kernel 2: fused GEMM (bf16x3 WMMA) + argmax + gather z_q + loss partials
//   grid = 512 blocks x 256 threads (8 waves); each block: 128 rows
// ---------------------------------------------------------------------------
__global__ __launch_bounds__(256) void vq_main_kernel(
        const float* __restrict__ z,
        const __bf16* __restrict__ eThi,
        const __bf16* __restrict__ eTlo,
        const float*  __restrict__ eTf,
        const float*  __restrict__ cn,
        float* __restrict__ out,
        float* __restrict__ partials) {

    __shared__ __bf16 sBhi[16 * LDS_B_STRIDE];
    __shared__ __bf16 sBlo[16 * LDS_B_STRIDE];
    __shared__ float  sCn[K_CB];
    __shared__ int    sIdx[8 * 16];
    __shared__ float  sLoss[8];

    const int tid  = threadIdx.x;
    const int wave = tid >> 5;
    const int lane = tid & 31;
    const int rowBase = blockIdx.x * 128 + wave * 16;

    // preload 0.5*||e||^2 into LDS (covered by barriers in the tile loop)
    for (int i = tid; i < K_CB; i += 256) sCn[i] = cn[i];

    // --- Build A fragments (bf16 hi/lo) in registers: 16 rows x 256, 8 k-steps
    // A layout (16-bit, 16x32): lanes 0-15 = rows, K 0..15 contiguous;
    // lanes 16-31 same rows, K 16..31 (matches B fill).
    const int aRow = rowBase + (lane & 15);
    const float* zr = z + (size_t)aRow * D_DIM;
    v16bf Ahi[8], Alo[8];
    #pragma unroll
    for (int s = 0; s < 8; ++s) {
        const int d0 = 32 * s + ((lane < 16) ? 0 : 16);
        const float4* zp = (const float4*)(zr + d0);
        float4 a0 = zp[0], a1 = zp[1];          // d0   .. d0+7
        float4 a2 = zp[2], a3 = zp[3];          // d0+8 .. d0+15
        float t[16] = { a0.x,a0.y,a0.z,a0.w, a1.x,a1.y,a1.z,a1.w,
                        a2.x,a2.y,a2.z,a2.w, a3.x,a3.y,a3.z,a3.w };
        AFrag hi, lo;
        #pragma unroll
        for (int i = 0; i < 16; ++i) {
            __bf16 h = (__bf16)t[i];
            hi.e[i] = h;
            lo.e[i] = (__bf16)(t[i] - (float)h);
        }
        Ahi[s] = hi.v;
        Alo[s] = lo.v;
    }

    float best[8];
    int   bidx[8];
    #pragma unroll
    for (int v = 0; v < 8; ++v) { best[v] = -1.0e30f; bidx[v] = 0; }

    const int ldR  = tid >> 4;          // cooperative B stage: code row 0..15
    const int ldCh = tid & 15;          // 32B chunk 0..15
    // per-thread LDS destinations (32-bit AS3 addresses) and global byte offset
    lds_void* ldsHi = (lds_void*)(sBhi + ldR * LDS_B_STRIDE + ldCh * 16);
    lds_void* ldsLo = (lds_void*)(sBlo + ldR * LDS_B_STRIDE + ldCh * 16);

    for (int t = 0; t < 64; ++t) {
        const int c0 = t * 16;
        __syncthreads();                // previous tile fully consumed
        {   // stage B tile (16 codes x 256 bf16, hi+lo) straight into LDS via
            // CDNA5 async-LDS loads (ASYNCcnt path, no VGPR round-trip).
            // NOTE: inst offset applies to BOTH the LDS and global address.
            const unsigned gboff =
                (unsigned)(((c0 + ldR) * D_DIM + ldCh * 16) * 2);  // bytes
            asm volatile(
                "global_load_async_to_lds_b128 %0, %2, %3 offset:0\n\t"
                "global_load_async_to_lds_b128 %0, %2, %3 offset:16\n\t"
                "global_load_async_to_lds_b128 %1, %2, %4 offset:0\n\t"
                "global_load_async_to_lds_b128 %1, %2, %4 offset:16"
                :
                : "v"(ldsHi), "v"(ldsLo), "v"(gboff), "s"(eThi), "s"(eTlo)
                : "memory");
            if (t + 1 < 64) {           // hint next tile toward L2/L0
                __builtin_prefetch(eThi + (size_t)(c0 + 16 + ldR) * D_DIM + ldCh * 16, 0, 1);
            }
            asm volatile("s_wait_asynccnt 0" ::: "memory");
        }
        __syncthreads();

        // --- 16x16 tile, three independent WMMA chains (breaks C-RAW chain):
        //     acc0 = Zhi*Ehi, acc1 = Zlo*Ehi, acc2 = Zhi*Elo (fp32 accum)
        v8f acc0 = {}, acc1 = {}, acc2 = {};
        const int cs = lane & 15;
        #pragma unroll
        for (int s = 0; s < 8; ++s) {
            const int d1 = 32 * s + ((lane < 16) ? 0 : 16);  // contiguous K, same as A
            BFrag bh, bl;
            const float4* ph = (const float4*)(sBhi + cs * LDS_B_STRIDE + d1);
            const float4* pl = (const float4*)(sBlo + cs * LDS_B_STRIDE + d1);
            bh.f4[0] = ph[0]; bh.f4[1] = ph[1];
            bl.f4[0] = pl[0]; bl.f4[1] = pl[1];
            acc0 = __builtin_amdgcn_wmma_f32_16x16x32_bf16(false, Ahi[s], false, bh.v,
                                                           (short)0, acc0, false, false);
            acc1 = __builtin_amdgcn_wmma_f32_16x16x32_bf16(false, Alo[s], false, bh.v,
                                                           (short)0, acc1, false, false);
            acc2 = __builtin_amdgcn_wmma_f32_16x16x32_bf16(false, Ahi[s], false, bl.v,
                                                           (short)0, acc2, false, false);
        }

        // running argmax of score = z.e - 0.5||e||^2 (== argmin distance);
        // codes increase with t, strict '>' keeps earliest index (ties -> min idx)
        const int c = c0 + cs;
        const float cneg = sCn[c];
        #pragma unroll
        for (int v = 0; v < 8; ++v) {
            float sc = (acc0[v] + acc1[v] + acc2[v]) - cneg;
            if (sc > best[v]) { best[v] = sc; bidx[v] = c; }
        }
    }

    // --- cross-lane argmax butterfly within each 16-lane half
    #pragma unroll
    for (int m = 1; m <= 8; m <<= 1) {
        #pragma unroll
        for (int v = 0; v < 8; ++v) {
            float os = __shfl_xor(best[v], m, 32);
            int   oi = __shfl_xor(bidx[v], m, 32);
            if (os > best[v] || (os == best[v] && oi < bidx[v])) {
                best[v] = os; bidx[v] = oi;
            }
        }
    }
    // lanes 0 / 16 hold rows v / v+8 (C/D layout)
    if ((lane & 15) == 0) {
        const int half = lane >> 4;
        #pragma unroll
        for (int v = 0; v < 8; ++v) sIdx[wave * 16 + half * 8 + v] = bidx[v];
    }
    __syncthreads();

    // --- z_q gather (exact fp32 code vectors), indices, loss partial
    float lossAcc = 0.0f;
    for (int r = 0; r < 16; ++r) {
        const int c   = sIdx[wave * 16 + r];
        const int row = rowBase + r;
        const float4* e4 = (const float4*)(eTf + (size_t)c * D_DIM + lane * 8);
        const float4* z4 = (const float4*)(z   + (size_t)row * D_DIM + lane * 8);
        float4 e0 = e4[0], e1 = e4[1];
        float4 z0 = z4[0], z1 = z4[1];
        float4* o4 = (float4*)(out + (size_t)row * D_DIM + lane * 8);
        o4[0] = e0; o4[1] = e1;
        float4 d0 = { e0.x - z0.x, e0.y - z0.y, e0.z - z0.z, e0.w - z0.w };
        float4 d1 = { e1.x - z1.x, e1.y - z1.y, e1.z - z1.z, e1.w - z1.w };
        lossAcc += d0.x*d0.x + d0.y*d0.y + d0.z*d0.z + d0.w*d0.w
                 + d1.x*d1.x + d1.y*d1.y + d1.z*d1.z + d1.w*d1.w;
        if (lane == 0) out[(size_t)N_ROWS * D_DIM + row] = (float)c;
    }

    // deterministic loss reduction: wave butterfly -> LDS -> one write per WG
    #pragma unroll
    for (int m = 16; m >= 1; m >>= 1) lossAcc += __shfl_xor(lossAcc, m, 32);
    if (lane == 0) sLoss[wave] = lossAcc;
    __syncthreads();
    if (tid == 0) {
        float s = 0.0f;
        #pragma unroll
        for (int w = 0; w < 8; ++w) s += sLoss[w];
        partials[blockIdx.x] = s;
    }
}

// ---------------------------------------------------------------------------
// Kernel 3: fixed-order reduction of 512 partials -> loss scalar
// ---------------------------------------------------------------------------
__global__ void vq_finalize_kernel(const float* __restrict__ partials,
                                   float* __restrict__ out) {
    __shared__ float red[256];
    int t = threadIdx.x;
    red[t] = partials[t] + partials[t + 256];
    __syncthreads();
    for (int w = 128; w > 0; w >>= 1) {
        if (t < w) red[t] += red[t + w];
        __syncthreads();
    }
    if (t == 0) {
        // loss = (1 + BETA) * mean((z_q - z)^2), BETA = 0.25
        out[(size_t)N_ROWS * D_DIM + N_ROWS] =
            red[0] * (1.25f / ((float)N_ROWS * (float)D_DIM));
    }
}

// ---------------------------------------------------------------------------
extern "C" void kernel_launch(void* const* d_in, const int* in_sizes, int n_in,
                              void* d_out, int out_size, void* d_ws, size_t ws_size,
                              hipStream_t stream) {
    const float* z = (const float*)d_in[0];          // [65536, 256]
    const float* E = (const float*)d_in[1];          // [256, 1024]
    float* out = (float*)d_out;                      // zq | idx | loss

    char* ws = (char*)d_ws;
    __bf16* eThi = (__bf16*)(ws);                    // 512 KB
    __bf16* eTlo = (__bf16*)(ws + (size_t)512 * 1024);
    float*  eTf  = (float*)(ws + (size_t)1024 * 1024);
    float*  cnrm = (float*)(ws + (size_t)2048 * 1024);
    float*  part = (float*)(ws + (size_t)2048 * 1024 + 4096);

    vq_prep_kernel<<<4, 256, 0, stream>>>(E, eThi, eTlo, eTf, cnrm);
    vq_main_kernel<<<512, 256, 0, stream>>>(z, eThi, eTlo, eTf, cnrm, out, part);
    vq_finalize_kernel<<<1, 256, 0, stream>>>(part, out);
}
